// AttentionBase_5669356832805
// MI455X (gfx1250) — compile-verified, matrix operand order checked by reading, not run on hardware
//
#include <hip/hip_runtime.h>
#include <hip/hip_bf16.h>

// Problem constants (match the reference)
#define B_   8
#define S_   1024
#define E_   1024
#define H_   16
#define D_   64     // head dim = E/H
#define KB   64     // k-rows processed per iteration
#define LDK  88     // LDS row stride (bf16) for K slab rows  (16B-aligned, bank-clean)
#define LDVT 88     // LDS row stride (bf16) for transposed V (16B-aligned, bank-clean)
#define LDSC 68     // LDS row stride (f32)  for score scratch (16B-aligned, bank-clean)
#define LDW  88     // LDS row stride (bf16) for W slab rows

typedef __attribute__((ext_vector_type(16))) __bf16 v16bf;
typedef __attribute__((ext_vector_type(8)))  float  v8f;

__device__ __forceinline__ v8f wmma_bf16(v16bf a, v16bf b, v8f c) {
    // D(f32 16x16) = A(bf16 16x32) x B(bf16 32x16) + C
    return __builtin_amdgcn_wmma_f32_16x16x32_bf16(
        /*neg_a=*/false, a, /*neg_b=*/false, b,
        /*c_mod=*/(short)0, c, /*reuse_a=*/false, /*reuse_b=*/false);
}

// ---------------------------------------------------------------------------
// Kernel 1: flash-attention. One block = 8 waves = 128 query rows of one (b,h)
// (halves K/V re-read traffic vs 64-row blocks). K-loop step = 64 rows.
// Writes the pre-projection attention output O (bf16, [B,S,E]) to workspace.
// ---------------------------------------------------------------------------
__global__ __launch_bounds__(256) void attn_kernel(
    const float* __restrict__ Kmat, const float* __restrict__ Vmat,
    const float* __restrict__ Qmat, const unsigned char* __restrict__ mask,
    __bf16* __restrict__ O)
{
    __shared__ __bf16 ldsK [KB * LDK];       // K slab, row-major [k][c]
    __shared__ __bf16 ldsVT[D_ * LDVT];      // V slab, TRANSPOSED [c][k]
    __shared__ float  sScr [8 * 16 * LDSC];  // per-wave f32 score scratch

    const int tid   = threadIdx.x;
    const int lane  = tid & 31;
    const int wave  = tid >> 5;
    const int lhalf = lane >> 4;        // 0: lanes 0-15, 1: lanes 16-31
    const int lmod  = lane & 15;
    const int c0    = lhalf * 16;       // K-offset within a 32-wide fragment

    const int b     = blockIdx.z;
    const int h     = blockIdx.y;
    const int q0    = blockIdx.x * 128 + wave * 16;
    const int hbase = h * D_;

    // --- Q A-fragments (16 rows x 64 dim -> two 16x32 fragments), pre-scaled
    v16bf aQ[2];
    {
        const int    row = q0 + lmod;
        const float* qp  = Qmat + ((size_t)b * S_ + row) * E_ + hbase;
        for (int f = 0; f < 2; ++f) {
            const float4* p4 = (const float4*)(qp + f * 32 + c0);
            #pragma unroll
            for (int j = 0; j < 4; ++j) {
                float4 v = p4[j];
                aQ[f][4*j+0] = (__bf16)(v.x * 0.125f);
                aQ[f][4*j+1] = (__bf16)(v.y * 0.125f);
                aQ[f][4*j+2] = (__bf16)(v.z * 0.125f);
                aQ[f][4*j+3] = (__bf16)(v.w * 0.125f);
            }
        }
    }

    v8f acc[4];
    #pragma unroll
    for (int ct = 0; ct < 4; ++ct)
        #pragma unroll
        for (int i = 0; i < 8; ++i) acc[ct][i] = 0.0f;

    float mrun = -3.0e38f;   // running row max  (A-layout: this lane's row = lmod)
    float lrun = 0.0f;       // running row sum

    float* scr = &sScr[wave * 16 * LDSC];

    for (int kb = 0; kb < S_; kb += KB) {
        // ---- cooperatively stage K (row-major) and V (transposed) as bf16
        {
            const int    r   = tid >> 2;          // 0..63  (k row)
            const int    cc  = (tid & 3) * 16;    // 0,16,32,48 (c chunk)
            const size_t gro = ((size_t)b * S_ + kb + r) * E_ + hbase + cc;
            const float4* kp = (const float4*)(Kmat + gro);
            const float4* vp = (const float4*)(Vmat + gro);
            #pragma unroll
            for (int j = 0; j < 4; ++j) {
                float4 kv = kp[j], vv = vp[j];
                ldsK[r*LDK + cc + 4*j+0] = (__bf16)kv.x;
                ldsK[r*LDK + cc + 4*j+1] = (__bf16)kv.y;
                ldsK[r*LDK + cc + 4*j+2] = (__bf16)kv.z;
                ldsK[r*LDK + cc + 4*j+3] = (__bf16)kv.w;
                // transpose on store: ldsVT[c][k]
                ldsVT[(cc + 4*j+0)*LDVT + r] = (__bf16)vv.x;
                ldsVT[(cc + 4*j+1)*LDVT + r] = (__bf16)vv.y;
                ldsVT[(cc + 4*j+2)*LDVT + r] = (__bf16)vv.z;
                ldsVT[(cc + 4*j+3)*LDVT + r] = (__bf16)vv.w;
            }
            if (kb + KB < S_) {   // global_prefetch_b8 of the next slab
                __builtin_prefetch(Kmat + gro + (size_t)KB * E_, 0, 3);
                __builtin_prefetch(Vmat + gro + (size_t)KB * E_, 0, 3);
            }
        }
        __syncthreads();

        // ---- S = Q K^T for this 16x64 score slab (four 16x16 N-tiles)
        v8f s[4];
        #pragma unroll
        for (int t = 0; t < 4; ++t)
            #pragma unroll
            for (int i = 0; i < 8; ++i) s[t][i] = 0.0f;
        #pragma unroll
        for (int t = 0; t < 4; ++t) {
            v16bf bK0, bK1;
            const __bf16* p0 = &ldsK[(t*16 + lmod) * LDK + c0];
            const __bf16* p1 = &ldsK[(t*16 + lmod) * LDK + 32 + c0];
            #pragma unroll
            for (int e = 0; e < 16; ++e) { bK0[e] = p0[e]; bK1[e] = p1[e]; }
            s[t] = wmma_bf16(aQ[0], bK0, s[t]);
            s[t] = wmma_bf16(aQ[1], bK1, s[t]);
        }

        // ---- additive mask bias (column = lane's N index in C-layout)
        float bias[4];
        #pragma unroll
        for (int t = 0; t < 4; ++t)
            bias[t] = mask[(size_t)b * S_ + kb + t*16 + lmod] ? 0.0f : -1.0e12f;

        // ---- C-layout -> A-layout transpose of biased scores via f32 scratch
        #pragma unroll
        for (int i = 0; i < 8; ++i) {
            const int r = i + 8 * lhalf;
            #pragma unroll
            for (int t = 0; t < 4; ++t)
                scr[r*LDSC + t*16 + lmod] = s[t][i] + bias[t];
        }
        float p[32];
        {
            const float4* r0 = (const float4*)&scr[lmod * LDSC + c0];
            const float4* r1 = (const float4*)&scr[lmod * LDSC + 32 + c0];
            #pragma unroll
            for (int j = 0; j < 4; ++j) {
                float4 a = r0[j], c = r1[j];
                p[4*j+0]  = a.x; p[4*j+1]  = a.y; p[4*j+2]  = a.z; p[4*j+3]  = a.w;
                p[16+4*j] = c.x; p[17+4*j] = c.y; p[18+4*j] = c.z; p[19+4*j] = c.w;
            }
        }

        // ---- online softmax, row-major: in-lane trees + one 16-xor shuffle
        float t16[16];
        #pragma unroll
        for (int e = 0; e < 16; ++e) t16[e] = fmaxf(p[e], p[e+16]);
        #pragma unroll
        for (int e = 0; e < 8; ++e)  t16[e] = fmaxf(t16[e], t16[e+8]);
        #pragma unroll
        for (int e = 0; e < 4; ++e)  t16[e] = fmaxf(t16[e], t16[e+4]);
        float bm = fmaxf(fmaxf(t16[0], t16[1]), fmaxf(t16[2], t16[3]));
        bm = fmaxf(bm, __shfl_xor(bm, 16, 32));          // other half of the row

        const float mnew = fmaxf(mrun, bm);
        const float cf   = __expf(mrun - mnew);
        mrun = mnew;

        float rs = 0.0f;
        #pragma unroll
        for (int e = 0; e < 32; ++e) { p[e] = __expf(p[e] - mnew); rs += p[e]; }
        rs += __shfl_xor(rs, 16, 32);
        lrun = lrun * cf + rs;

        v16bf aP0, aP1;
        #pragma unroll
        for (int e = 0; e < 16; ++e) { aP0[e] = (__bf16)p[e]; aP1[e] = (__bf16)p[e+16]; }

        // broadcast cf back to C-layout rows (row of VGPR i is i + 8*lhalf)
        float cfc[8];
        #pragma unroll
        for (int i = 0; i < 8; ++i) cfc[i] = __shfl(cf, i + 8*lhalf, 32);
        #pragma unroll
        for (int ct = 0; ct < 4; ++ct)
            #pragma unroll
            for (int i = 0; i < 8; ++i) acc[ct][i] *= cfc[i];

        // ---- P·V: B-fragments are CONTIGUOUS reads from transposed V
        #pragma unroll
        for (int ct = 0; ct < 4; ++ct) {
            v16bf bV0, bV1;
            const __bf16* v0 = &ldsVT[(ct*16 + lmod) * LDVT + c0];
            const __bf16* v1 = &ldsVT[(ct*16 + lmod) * LDVT + 32 + c0];
            #pragma unroll
            for (int e = 0; e < 16; ++e) { bV0[e] = v0[e]; bV1[e] = v1[e]; }
            acc[ct] = wmma_bf16(aP0, bV0, acc[ct]);
            acc[ct] = wmma_bf16(aP1, bV1, acc[ct]);
        }
        __syncthreads();
    }

    // ---- epilogue: normalize and write O as bf16 (halves intermediate traffic)
    const float linv = 1.0f / lrun;
    float lc[8];
    #pragma unroll
    for (int i = 0; i < 8; ++i) lc[i] = __shfl(linv, i + 8*lhalf, 32);
    #pragma unroll
    for (int ct = 0; ct < 4; ++ct)
        #pragma unroll
        for (int i = 0; i < 8; ++i) {
            const int row = q0 + i + 8 * lhalf;
            const int col = hbase + ct*16 + lmod;
            O[((size_t)b * S_ + row) * E_ + col] = (__bf16)(acc[ct][i] * lc[i]);
        }
}

// ---------------------------------------------------------------------------
// Kernel 2: Y[m,n] = sum_f O[m,f] * W[n,f]   (out @ w_out.T), O is bf16
// Block = 128x64 output tile (8 waves on M), 64-deep K-slabs of W via LDS.
// ---------------------------------------------------------------------------
__global__ __launch_bounds__(256) void proj_kernel(
    const __bf16* __restrict__ O, const float* __restrict__ W,
    float* __restrict__ Y)
{
    __shared__ __bf16 ldsW[64 * LDW];

    const int tid   = threadIdx.x;
    const int lane  = tid & 31;
    const int wave  = tid >> 5;
    const int lhalf = lane >> 4;
    const int lmod  = lane & 15;
    const int c0    = lhalf * 16;

    const int m0 = blockIdx.x * 128 + wave * 16;
    const int n0 = blockIdx.y * 64;

    v8f acc[4];
    #pragma unroll
    for (int ct = 0; ct < 4; ++ct)
        #pragma unroll
        for (int i = 0; i < 8; ++i) acc[ct][i] = 0.0f;

    for (int kc = 0; kc < E_; kc += 64) {
        // stage W[n0..n0+63][kc..kc+63] as bf16 (rows are K-contiguous:
        // exactly the B-fragment order since we multiply by W^T)
        {
            const int r  = tid >> 2;          // 0..63
            const int cc = (tid & 3) * 16;    // 0,16,32,48
            const float4* wp = (const float4*)(W + (size_t)(n0 + r) * E_ + kc + cc);
            #pragma unroll
            for (int j = 0; j < 4; ++j) {
                float4 wv = wp[j];
                ldsW[r*LDW + cc + 4*j+0] = (__bf16)wv.x;
                ldsW[r*LDW + cc + 4*j+1] = (__bf16)wv.y;
                ldsW[r*LDW + cc + 4*j+2] = (__bf16)wv.z;
                ldsW[r*LDW + cc + 4*j+3] = (__bf16)wv.w;
            }
            if (kc + 64 < E_)
                __builtin_prefetch(W + (size_t)(n0 + r) * E_ + kc + 64 + cc, 0, 3);
        }
        __syncthreads();

        // A fragments straight from the bf16 intermediate: no conversion VALU
        v16bf aO0, aO1;
        {
            const __bf16* op = O + (size_t)(m0 + lmod) * E_ + kc;
            #pragma unroll
            for (int e = 0; e < 16; ++e) {
                aO0[e] = op[c0 + e];
                aO1[e] = op[32 + c0 + e];
            }
        }
        #pragma unroll
        for (int ct = 0; ct < 4; ++ct) {
            v16bf bW0, bW1;
            const __bf16* p0 = &ldsW[(ct*16 + lmod) * LDW + c0];
            const __bf16* p1 = &ldsW[(ct*16 + lmod) * LDW + 32 + c0];
            #pragma unroll
            for (int e = 0; e < 16; ++e) { bW0[e] = p0[e]; bW1[e] = p1[e]; }
            acc[ct] = wmma_bf16(aO0, bW0, acc[ct]);
            acc[ct] = wmma_bf16(aO1, bW1, acc[ct]);
        }
        __syncthreads();
    }

    #pragma unroll
    for (int ct = 0; ct < 4; ++ct)
        #pragma unroll
        for (int i = 0; i < 8; ++i) {
            const int m = m0 + i + 8 * lhalf;
            const int n = n0 + ct*16 + lmod;
            Y[(size_t)m * E_ + n] = acc[ct][i];
        }
}

// ---------------------------------------------------------------------------
extern "C" void kernel_launch(void* const* d_in, const int* in_sizes, int n_in,
                              void* d_out, int out_size, void* d_ws, size_t ws_size,
                              hipStream_t stream) {
    const float*         keys    = (const float*)d_in[0];
    const float*         values  = (const float*)d_in[1];
    const float*         queries = (const float*)d_in[2];
    const unsigned char* amask   = (const unsigned char*)d_in[3];
    const float*         w_out   = (const float*)d_in[4];
    __bf16* O = (__bf16*)d_ws;   // intermediate attention output [B,S,E] bf16
    float*  Y = (float*)d_out;

    attn_kernel<<<dim3(S_ / 128, H_, B_), 256, 0, stream>>>(
        keys, values, queries, amask, O);
    proj_kernel<<<dim3((B_ * S_) / 128, E_ / 64), 256, 0, stream>>>(
        O, w_out, Y);
}
